// LSTMCell_54614804136122
// MI455X (gfx1250) — matrix operand order repaired and checked by reading, not compiled, not run on hardware
//
#include <hip/hip_runtime.h>
#include <hip/hip_bf16.h>
#include <math.h>

typedef __attribute__((ext_vector_type(16))) _Float16 v16h;
typedef __attribute__((ext_vector_type(8)))  _Float16 v8h;
typedef __attribute__((ext_vector_type(8)))  float    v8f;
typedef __attribute__((ext_vector_type(4)))  float    f32x4;

#define TILE_M 64      // batch rows per block
#define TILE_N 64      // hidden cols per block (per gate; x4 gates computed)
#define TILE_K 32      // K per WMMA step
#define LDS_LD 40      // padded row stride in halves (32 + 8) -> 80B, conflict-free b128 reads

__device__ __forceinline__ v8h cvt8_f32_to_f16(const float* __restrict__ p) {
    f32x4 f0 = *(const f32x4*)p;
    f32x4 f1 = *(const f32x4*)(p + 4);
    v8h h;
    h[0] = (_Float16)f0[0]; h[1] = (_Float16)f0[1];
    h[2] = (_Float16)f0[2]; h[3] = (_Float16)f0[3];
    h[4] = (_Float16)f1[0]; h[5] = (_Float16)f1[1];
    h[6] = (_Float16)f1[2]; h[7] = (_Float16)f1[3];
    return h;
}

__device__ __forceinline__ float sigmoidf_fast(float x) {
    return 1.0f / (1.0f + __expf(-x));
}

__global__ __launch_bounds__(256, 2)
void lstm_cell_wmma_f16(const float* __restrict__ input,    // [4096,1024]
                        const float* __restrict__ hx,       // [4096,1024]
                        const float* __restrict__ cx,       // [4096,1024]
                        const float* __restrict__ noise_q,  // [1]
                        const float* __restrict__ noise_e,  // [1]
                        const float* __restrict__ w_ih,     // [4096,1024]
                        const float* __restrict__ w_hh,     // [4096,1024]
                        const float* __restrict__ b_ih,     // [4096]
                        const float* __restrict__ b_hh,     // [4096]
                        const float* __restrict__ eps_c,    // [4096,1024]
                        const float* __restrict__ eps_h,    // [4096,1024]
                        float* __restrict__ out)            // hy[4096*1024] ++ cy[4096*1024]
{
    constexpr int Bdim = 4096, Idim = 1024, Hdim = 1024, Ktot = 2048;

    // A tile: 64 rows x 32 k ; B tile: 4 gates x 64 W-rows x 32 k
    __shared__ _Float16 ldsA[TILE_M * LDS_LD];          //  5 KB
    __shared__ _Float16 ldsB[4 * TILE_N * LDS_LD];      // 20 KB

    const int tid  = threadIdx.x;
    const int lane = tid & 31;
    const int wave = tid >> 5;       // 0..7
    const int wm   = wave >> 2;      // 0..1  : 32-row sub-block of M
    const int wn   = wave & 3;       // 0..3  : 16-col sub-block of N

    const int blockM = blockIdx.x * TILE_M;   // batch base
    const int blockN = blockIdx.y * TILE_N;   // hidden base

    // acc[gate][mi] : 16x16 f32 C/D tile, lane layout per ISA 7.12.2
    v8f acc[4][2];
#pragma unroll
    for (int g = 0; g < 4; ++g)
#pragma unroll
        for (int mi = 0; mi < 2; ++mi)
            acc[g][mi] = (v8f)(0.0f);

    // staging assignments (256 threads)
    const int a_row = tid >> 2;            // 0..63
    const int a_kq  = (tid & 3) * 8;       // 0,8,16,24
    const int b_row = tid;                 // 0..255 : gate = b_row>>6, n = b_row&63
    const int b_gate = b_row >> 6;
    const int b_n    = b_row & 63;

    for (int k0 = 0; k0 < Ktot; k0 += TILE_K) {
        const bool  first = (k0 < Idim);
        const float* Asrc = first ? input : hx;
        const float* Bsrc = first ? w_ih  : w_hh;
        const int    ksrc = first ? k0 : (k0 - Idim);

        __syncthreads();   // protect LDS reads of previous iteration

        // ---- stage A: 64x32 fp32 -> f16 LDS ----
        {
            const float* p = Asrc + (size_t)(blockM + a_row) * Idim + ksrc + a_kq;
            *(v8h*)&ldsA[a_row * LDS_LD + a_kq] = cvt8_f32_to_f16(p);
        }
        // ---- stage B: 256x32 fp32 -> f16 LDS (W rows for 4 gates) ----
        {
            const float* p = Bsrc + (size_t)(b_gate * Hdim + blockN + b_n) * Idim + ksrc;
#pragma unroll
            for (int q = 0; q < 4; ++q)
                *(v8h*)&ldsB[b_row * LDS_LD + q * 8] = cvt8_f32_to_f16(p + q * 8);
        }

        __syncthreads();

        // ---- fragments per ISA VGPR layouts (wave32) ----
        const int r16  = lane & 15;
        const int c0   = (lane < 16) ? 0 : 8;    // A: k-half select
        const int kb   = (lane < 16) ? 0 : 8;    // B: k-half select

        v16h afrag[2];
#pragma unroll
        for (int mi = 0; mi < 2; ++mi) {
            const _Float16* base = &ldsA[(wm * 32 + mi * 16 + r16) * LDS_LD];
            v8h lo = *(const v8h*)(base + c0);        // K = c0..c0+7
            v8h hi = *(const v8h*)(base + 16 + c0);   // K = 16+c0..16+c0+7
            afrag[mi] = __builtin_shufflevector(lo, hi,
                0, 1, 2, 3, 4, 5, 6, 7, 8, 9, 10, 11, 12, 13, 14, 15);
        }

#pragma unroll
        for (int g = 0; g < 4; ++g) {
            const _Float16* base = &ldsB[(g * 64 + wn * 16 + r16) * LDS_LD + kb];
            v8h lo = *(const v8h*)(base);         // K = kb..kb+7
            v8h hi = *(const v8h*)(base + 16);    // K = 16+kb..16+kb+7
            v16h bfrag = __builtin_shufflevector(lo, hi,
                0, 1, 2, 3, 4, 5, 6, 7, 8, 9, 10, 11, 12, 13, 14, 15);
#pragma unroll
            for (int mi = 0; mi < 2; ++mi) {
                acc[g][mi] = __builtin_amdgcn_wmma_f32_16x16x32_f16(
                    false, afrag[mi], false, bfrag,
                    (short)0, acc[g][mi], false, false);
            }
        }
    }

    // ---- fused epilogue: all 4 gates for this (b,h) patch are in-register ----
    const int m16 = (lane >> 4) * 8;       // D-tile: lanes>=16 hold M=8..15
    const int nc  = lane & 15;
    const int h_g = blockN + wn * 16 + nc;

    float bias[4];
#pragma unroll
    for (int g = 0; g < 4; ++g)
        bias[g] = b_ih[g * Hdim + h_g] + b_hh[g * Hdim + h_g];

    const float se = sqrtf(noise_e[0]);
    const float sq = sqrtf(noise_q[0]);

#pragma unroll
    for (int mi = 0; mi < 2; ++mi) {
#pragma unroll
        for (int r = 0; r < 8; ++r) {
            const int m_g = blockM + wm * 32 + mi * 16 + m16 + r;
            const size_t idx = (size_t)m_g * Hdim + h_g;

            const float ig = sigmoidf_fast(acc[0][mi][r] + bias[0]);
            const float fg = sigmoidf_fast(acc[1][mi][r] + bias[1]);
            const float cg = tanhf        (acc[2][mi][r] + bias[2]);
            const float og = sigmoidf_fast(acc[3][mi][r] + bias[3]);

            const float cy = fg * cx[idx] + ig * cg + se * eps_c[idx];
            const float hy = og * tanhf(cy) + sq * eps_h[idx];

            out[idx]                          = hy;   // hy first
            out[(size_t)Bdim * Hdim + idx]    = cy;   // then cy
        }
    }
}

extern "C" void kernel_launch(void* const* d_in, const int* in_sizes, int n_in,
                              void* d_out, int out_size, void* d_ws, size_t ws_size,
                              hipStream_t stream) {
    (void)in_sizes; (void)n_in; (void)out_size; (void)d_ws; (void)ws_size;

    const float* input   = (const float*)d_in[0];
    const float* hx      = (const float*)d_in[1];
    const float* cx      = (const float*)d_in[2];
    const float* noise_q = (const float*)d_in[3];
    const float* noise_e = (const float*)d_in[4];
    const float* w_ih    = (const float*)d_in[5];
    const float* w_hh    = (const float*)d_in[6];
    const float* b_ih    = (const float*)d_in[7];
    const float* b_hh    = (const float*)d_in[8];
    const float* eps_c   = (const float*)d_in[9];
    const float* eps_h   = (const float*)d_in[10];
    float* out = (float*)d_out;

    dim3 grid(4096 / TILE_M, 1024 / TILE_N);   // 64 x 16 blocks
    dim3 block(256);                            // 8 waves (wave32)
    lstm_cell_wmma_f16<<<grid, block, 0, stream>>>(
        input, hx, cx, noise_q, noise_e,
        w_ih, w_hh, b_ih, b_hh, eps_c, eps_h, out);
}
